// DTDMN_83829171683410
// MI455X (gfx1250) — compile-verified, matrix operand order checked by reading, not run on hardware
//
#include <hip/hip_runtime.h>
#include <hip/hip_bf16.h>

// ---------------- problem constants (from reference) ----------------
#define NSENT 1024
#define NWORD 40
#define EDIM  300
#define EPAD  320      // E padded to multiple of 32
#define HDIM  512
#define VDIM  20000
#define DDIM  50
#define KDIM  50
#define NPAD64 64      // D/K padded to 64
#define DKPAD 128      // concatenated discourse+topic K, padded
#define VPAD  20032    // V padded to multiple of 64
#define MDIM  256
#define BDLG  32
#define DSAMP 5

typedef __attribute__((ext_vector_type(16))) _Float16 v16h;
typedef __attribute__((ext_vector_type(8)))  _Float16 v8h;
typedef __attribute__((ext_vector_type(8)))  float    v8f;

// ---- WMMA wrapper: D = A(16x32 f16) * B(32x16 f16) + C(f32) ----
__device__ __forceinline__ v8f wmma_ff(v16h a, v16h b, v8f c) {
    return __builtin_amdgcn_wmma_f32_16x16x32_f16(
        /*neg_a=*/false, a, /*neg_b=*/false, b,
        /*c_mod=*/(short)0, c, /*reuse_a=*/false, /*reuse_b=*/false);
}

// ---- CDNA5 async DMA: global -> LDS, 16B per lane, tracked by ASYNCcnt ----
__device__ __forceinline__ void async_g2l_b128(const void* g, void* l) {
    unsigned loff = (unsigned)(size_t)l;                 // LDS aperture: addr[31:0]
    unsigned long long ga = (unsigned long long)(size_t)g;
    asm volatile("global_load_async_to_lds_b128 %0, %1, off"
                 :: "v"(loff), "v"(ga) : "memory");
}
// in-order per-wave ASYNCcnt waits (double-buffer pipeline)
__device__ __forceinline__ void wait_async_le0() { asm volatile("s_wait_asynccnt 0x0" ::: "memory"); }
__device__ __forceinline__ void wait_async_le2() { asm volatile("s_wait_asynccnt 0x2" ::: "memory"); }
__device__ __forceinline__ void wait_async_le4() { asm volatile("s_wait_asynccnt 0x4" ::: "memory"); }

// Fragment gather from an LDS tile stored [row][40 halves] (80B stride, 16B
// aligned).  Lane l<16: row l, K={0..7,16..23}; lane l>=16: row l-16,
// K={8..15,24..31}  (ISA 7.12.2 16-bit A/B layout).  Two ds_load_b128.
__device__ __forceinline__ v16h frag_lds(const _Float16* rowp, int lane) {
    const int khi = (lane & 16) ? 8 : 0;
    v8h lo = *(const v8h*)(rowp + khi);
    v8h hi = *(const v8h*)(rowp + 16 + khi);
    v16h r;
#pragma unroll
    for (int j = 0; j < 8; ++j) { r[j] = lo[j]; r[j + 8] = hi[j]; }
    return r;
}

__device__ __forceinline__ float act_apply(float x, int act) {
    if (act == 1) return tanhf(x);
    if (act == 2) return 1.0f / (1.0f + __expf(-x));
    return x;
}

__device__ __forceinline__ unsigned hash_u32(unsigned x) {
    x ^= x >> 16; x *= 0x7feb352dU;
    x ^= x >> 15; x *= 0x846ca68bU;
    x ^= x >> 16; return x;
}
__device__ __forceinline__ float u01(unsigned x) {
    return (float)(hash_u32(x) >> 8) * (1.0f / 16777216.0f) + 1e-7f;
}

// ---------------- utility kernels ----------------
__global__ void zero_f32_kernel(float* p, int n) {
    int i = blockIdx.x * blockDim.x + threadIdx.x;
    if (i < n) p[i] = 0.0f;
}
__global__ void add_bias_kernel(const float* __restrict__ a, const float* __restrict__ b,
                                float* __restrict__ o, int n) {
    int i = blockIdx.x * blockDim.x + threadIdx.x;
    if (i < n) o[i] = a[i] + b[i];
}

// f32 [srows, scols] -> f16 block at dst[r*dld + dcol0 + c], zero padded over
// [drows, dcols].
__global__ void cvt2d_f16_kernel(const float* __restrict__ src, _Float16* __restrict__ dst,
                                 int srows, int scols, int drows, int dcols,
                                 int dld, int dcol0) {
    int i = blockIdx.x * blockDim.x + threadIdx.x;
    if (i >= drows * dcols) return;
    int r = i / dcols, c = i - r * dcols;
    float v = (r < srows && c < scols) ? src[(size_t)r * scols + c] : 0.0f;
    dst[(size_t)r * dld + dcol0 + c] = (_Float16)v;
}

// ---------------- generic tiled WMMA GEMM (all-f16, async double-buffered) ----
// C[M,N] = act( A[M,K] * B[N,K]^T + bias[N] )
// REQUIREMENTS: M % 64 == 0, K % 32 == 0, lda/ldb % 8 == 0, B has >= ceil64(N)
// zero-padded rows.  Tile k+1 DMAs overlap tile k WMMAs (ASYNCcnt <= 2).
__global__ __launch_bounds__(256)
void gemm_wmma_kernel(const _Float16* __restrict__ A, int lda,
                      const _Float16* __restrict__ B, int ldb,
                      const float* __restrict__ bias,
                      float* __restrict__ C, _Float16* __restrict__ C16, int ldc,
                      int Nd, int Kd, int act, int nt) {
    __shared__ __align__(16) _Float16 As[2][64][40];
    __shared__ __align__(16) _Float16 Bs[2][64][40];

    const int tid  = threadIdx.x;
    const int lane = tid & 31;
    const int wid  = tid >> 5;
    const int m0   = blockIdx.y * 64;
    const int n0   = blockIdx.x * 64;
    const int row0 = (wid & 3) << 4;
    const int col0 = (wid >> 2) << 5;
    const int lr   = tid >> 2;        // 0..63
    const int lk   = (tid & 3) << 3;  // 0,8,16,24

    v8f acc0 = {0.f,0.f,0.f,0.f,0.f,0.f,0.f,0.f};
    v8f acc1 = acc0;

    const _Float16* pa = A + (size_t)(m0 + lr) * lda + lk;
    const _Float16* pb = B + (size_t)(n0 + lr) * ldb + lk;

    async_g2l_b128(pa, &As[0][lr][lk]);        // prologue: tile 0
    async_g2l_b128(pb, &Bs[0][lr][lk]);
    for (int k0 = 0; k0 < Kd; k0 += 32) {
        const int cur = (k0 >> 5) & 1;
        if (k0 + 32 < Kd) {                    // stream tile k+1 while computing k
            async_g2l_b128(pa + k0 + 32, &As[cur ^ 1][lr][lk]);
            async_g2l_b128(pb + k0 + 32, &Bs[cur ^ 1][lr][lk]);
            wait_async_le2();
        } else {
            wait_async_le0();
        }
        __syncthreads();
        v16h af = frag_lds(&As[cur][row0 + (lane & 15)][0], lane);
        v16h b0 = frag_lds(&Bs[cur][col0 + (lane & 15)][0], lane);
        v16h b1 = frag_lds(&Bs[cur][col0 + 16 + (lane & 15)][0], lane);
        acc0 = wmma_ff(af, b0, acc0);
        acc1 = wmma_ff(af, b1, acc1);
        __syncthreads();
    }

    const int rbase = m0 + row0 + ((lane & 16) ? 8 : 0);
    const int c0 = n0 + col0 + (lane & 15);
    const int c1 = c0 + 16;
#pragma unroll
    for (int v = 0; v < 8; ++v) {
        int r = rbase + v;
        if (c0 < Nd) {
            float x = act_apply(acc0[v] + (bias ? bias[c0] : 0.0f), act);
            size_t o = (size_t)r * ldc + c0;
            if (nt) __builtin_nontemporal_store(x, &C[o]); else C[o] = x;
            if (C16) C16[o] = (_Float16)x;
        }
        if (c1 < Nd) {
            float x = act_apply(acc1[v] + (bias ? bias[c1] : 0.0f), act);
            size_t o = (size_t)r * ldc + c1;
            if (nt) __builtin_nontemporal_store(x, &C[o]); else C[o] = x;
            if (C16) C16[o] = (_Float16)x;
        }
    }
}

// ---------------- fused GRU timestep (async double-buffered) ----------------
// gates = emb_t(word gather, K padded to 320) @ w_ih^T + h @ w_hh^T, gi/gh
// kept separate (n-gate needs r*gh_n), sigmoid/tanh + length mask in epilogue.
__global__ __launch_bounds__(256)
void gru_step_kernel(const int* __restrict__ word_id,
                     const _Float16* __restrict__ wemb,  // [V, EPAD] f16
                     const _Float16* __restrict__ wih,   // [3H, EPAD] f16
                     const _Float16* __restrict__ whh,   // [3H, H] f16
                     const float* __restrict__ b_ih, const float* __restrict__ b_hh,
                     const int* __restrict__ sent_lens,
                     float* __restrict__ h32, _Float16* __restrict__ h16,
                     _Float16* __restrict__ out_h16, int t) {
    __shared__ __align__(16) _Float16 As[2][64][40];
    __shared__ __align__(16) _Float16 Bs[2][3][64][40];

    const int tid  = threadIdx.x;
    const int lane = tid & 31;
    const int wid  = tid >> 5;
    const int n0   = blockIdx.y * 64;   // sentence base
    const int j0   = blockIdx.x * 64;   // hidden-col base
    const int row0 = (wid & 3) << 4;
    const int col0 = (wid >> 2) << 5;
    const int lr   = tid >> 2;
    const int lk   = (tid & 3) << 3;

    v8f acc[3][2][2];
#pragma unroll
    for (int g = 0; g < 3; ++g)
#pragma unroll
        for (int p = 0; p < 2; ++p)
#pragma unroll
            for (int c = 0; c < 2; ++c)
                acc[g][p][c] = (v8f){0.f,0.f,0.f,0.f,0.f,0.f,0.f,0.f};

    const int widx = word_id[(n0 + lr) * NWORD + t];
    const _Float16* pea = wemb + (size_t)widx * EPAD + lk;

    // ---- phase 1: K over EPAD (input projection, gathered embedding) ----
    async_g2l_b128(pea, &As[0][lr][lk]);
#pragma unroll
    for (int g = 0; g < 3; ++g)
        async_g2l_b128(wih + (size_t)(g * HDIM + j0 + lr) * EPAD + lk, &Bs[0][g][lr][lk]);
    for (int k0 = 0; k0 < EPAD; k0 += 32) {
        const int cur = (k0 >> 5) & 1;
        if (k0 + 32 < EPAD) {
            async_g2l_b128(pea + k0 + 32, &As[cur ^ 1][lr][lk]);
#pragma unroll
            for (int g = 0; g < 3; ++g)
                async_g2l_b128(wih + (size_t)(g * HDIM + j0 + lr) * EPAD + k0 + 32 + lk,
                               &Bs[cur ^ 1][g][lr][lk]);
            wait_async_le4();
        } else {
            wait_async_le0();
        }
        __syncthreads();
        v16h af = frag_lds(&As[cur][row0 + (lane & 15)][0], lane);
#pragma unroll
        for (int g = 0; g < 3; ++g)
#pragma unroll
            for (int cc = 0; cc < 2; ++cc) {
                v16h bf = frag_lds(&Bs[cur][g][col0 + cc * 16 + (lane & 15)][0], lane);
                acc[g][0][cc] = wmma_ff(af, bf, acc[g][0][cc]);
            }
        __syncthreads();
    }

    // ---- phase 2: K over H (recurrent projection) ----
    const _Float16* pha = h16 + (size_t)(n0 + lr) * HDIM + lk;
    async_g2l_b128(pha, &As[0][lr][lk]);
#pragma unroll
    for (int g = 0; g < 3; ++g)
        async_g2l_b128(whh + (size_t)(g * HDIM + j0 + lr) * HDIM + lk, &Bs[0][g][lr][lk]);
    for (int k0 = 0; k0 < HDIM; k0 += 32) {
        const int cur = (k0 >> 5) & 1;
        if (k0 + 32 < HDIM) {
            async_g2l_b128(pha + k0 + 32, &As[cur ^ 1][lr][lk]);
#pragma unroll
            for (int g = 0; g < 3; ++g)
                async_g2l_b128(whh + (size_t)(g * HDIM + j0 + lr) * HDIM + k0 + 32 + lk,
                               &Bs[cur ^ 1][g][lr][lk]);
            wait_async_le4();
        } else {
            wait_async_le0();
        }
        __syncthreads();
        v16h af = frag_lds(&As[cur][row0 + (lane & 15)][0], lane);
#pragma unroll
        for (int g = 0; g < 3; ++g)
#pragma unroll
            for (int cc = 0; cc < 2; ++cc) {
                v16h bf = frag_lds(&Bs[cur][g][col0 + cc * 16 + (lane & 15)][0], lane);
                acc[g][1][cc] = wmma_ff(af, bf, acc[g][1][cc]);
            }
        __syncthreads();
    }

    // ---- epilogue: gate math + masked update ----
    const int rbase = row0 + ((lane & 16) ? 8 : 0);
#pragma unroll
    for (int cc = 0; cc < 2; ++cc) {
        const int j = j0 + col0 + cc * 16 + (lane & 15);
#pragma unroll
        for (int v = 0; v < 8; ++v) {
            const int n = n0 + rbase + v;
            float ir = acc[0][0][cc][v] + b_ih[j];
            float hr = acc[0][1][cc][v] + b_hh[j];
            float iz = acc[1][0][cc][v] + b_ih[HDIM + j];
            float hz = acc[1][1][cc][v] + b_hh[HDIM + j];
            float inn = acc[2][0][cc][v] + b_ih[2 * HDIM + j];
            float hn  = acc[2][1][cc][v] + b_hh[2 * HDIM + j];
            float r = 1.0f / (1.0f + __expf(-(ir + hr)));
            float z = 1.0f / (1.0f + __expf(-(iz + hz)));
            float ng = tanhf(inn + r * hn);
            float hp = h32[(size_t)n * HDIM + j];
            float hv = (t < sent_lens[n]) ? ((1.0f - z) * ng + z * hp) : hp;
            h32[(size_t)n * HDIM + j] = hv;
            _Float16 hh = (_Float16)hv;
            h16[(size_t)n * HDIM + j] = hh;
            out_h16[((size_t)n * NWORD + t) * HDIM + j] = hh;
        }
    }
}

// -------- attention scores: s[n,t] = v . tanh(h W^T + b), N looped in-kernel --------
__global__ __launch_bounds__(256)
void attn_score_kernel(const _Float16* __restrict__ hdat,   // [40960, 512] f16
                       const _Float16* __restrict__ attnW,  // [512, 512] f16
                       const float* __restrict__ attn_b,
                       const float* __restrict__ attn_v,
                       float* __restrict__ scores) {        // [40960]
    __shared__ __align__(16) _Float16 As[2][64][40];
    __shared__ __align__(16) _Float16 Bs[2][64][40];
    __shared__ float sred[64];

    const int tid  = threadIdx.x;
    const int lane = tid & 31;
    const int wid  = tid >> 5;
    const int m0   = blockIdx.x * 64;
    const int row0 = (wid & 3) << 4;
    const int col0 = (wid >> 2) << 5;
    const int lr   = tid >> 2;
    const int lk   = (tid & 3) << 3;

    if (tid < 64) sred[tid] = 0.0f;

    const _Float16* pa = hdat + (size_t)(m0 + lr) * HDIM + lk;

    for (int n0 = 0; n0 < HDIM; n0 += 64) {
        v8f acc0 = {0.f,0.f,0.f,0.f,0.f,0.f,0.f,0.f};
        v8f acc1 = acc0;
        const _Float16* pb = attnW + (size_t)(n0 + lr) * HDIM + lk;
        async_g2l_b128(pa, &As[0][lr][lk]);
        async_g2l_b128(pb, &Bs[0][lr][lk]);
        for (int k0 = 0; k0 < HDIM; k0 += 32) {
            const int cur = (k0 >> 5) & 1;
            if (k0 + 32 < HDIM) {
                async_g2l_b128(pa + k0 + 32, &As[cur ^ 1][lr][lk]);
                async_g2l_b128(pb + k0 + 32, &Bs[cur ^ 1][lr][lk]);
                wait_async_le2();
            } else {
                wait_async_le0();
            }
            __syncthreads();
            v16h af = frag_lds(&As[cur][row0 + (lane & 15)][0], lane);
            v16h b0 = frag_lds(&Bs[cur][col0 + (lane & 15)][0], lane);
            v16h b1 = frag_lds(&Bs[cur][col0 + 16 + (lane & 15)][0], lane);
            acc0 = wmma_ff(af, b0, acc0);
            acc1 = wmma_ff(af, b1, acc1);
            __syncthreads();
        }
        const int rl = row0 + ((lane & 16) ? 8 : 0);
        const int c0 = n0 + col0 + (lane & 15);
        const int c1 = c0 + 16;
#pragma unroll
        for (int v = 0; v < 8; ++v) {
            float p = tanhf(acc0[v] + attn_b[c0]) * attn_v[c0]
                    + tanhf(acc1[v] + attn_b[c1]) * attn_v[c1];
            atomicAdd(&sred[rl + v], p);   // ds_add_f32
        }
        __syncthreads();
    }
    if (tid < 64) scores[m0 + tid] = sred[tid];
}

// ---------------- masked softmax over T + attention pooling ----------------
__global__ void softmax_pool_kernel(const float* __restrict__ scores,
                                    const int* __restrict__ sent_lens,
                                    const _Float16* __restrict__ hdat,
                                    float* __restrict__ sent_emb) {
    const int n = blockIdx.x;
    const int tid = threadIdx.x;  // 128
    __shared__ float sa[NWORD];
    __shared__ float red;
    const int L = sent_lens[n];
    if (tid < NWORD) sa[tid] = (tid < L) ? scores[n * NWORD + tid] : -1e30f;
    __syncthreads();
    if (tid == 0) {
        float m = -1e30f;
        for (int t = 0; t < NWORD; ++t) m = fmaxf(m, sa[t]);
        red = m;
    }
    __syncthreads();
    float mx = red;
    if (tid < NWORD) sa[tid] = (tid < L) ? __expf(sa[tid] - mx) : 0.0f;
    __syncthreads();
    if (tid == 0) {
        float s = 0.0f;
        for (int t = 0; t < NWORD; ++t) s += sa[t];
        red = s;
    }
    __syncthreads();
    const float inv = 1.0f / red;
    for (int j = tid; j < HDIM; j += 128) {
        float a = 0.0f;
        for (int t = 0; t < NWORD; ++t)
            a += sa[t] * (float)hdat[((size_t)n * NWORD + t) * HDIM + j];
        sent_emb[(size_t)n * HDIM + j] = a * inv;
    }
}

// ------ gumbel-softmax mean; writes f32 + f16 slice of fused decoder A ------
__global__ void gumbel_kernel(const float* __restrict__ qd,
                              float* __restrict__ sample_d,
                              _Float16* __restrict__ dfuse16) {  // [NS, 128], cols 0..63
    const int n = blockIdx.x, d = threadIdx.x;  // 64 threads
    __shared__ float sv[64];
    __shared__ float red;
    const float base = (d < DDIM) ? qd[n * DDIM + d] : 0.0f;
    float accp = 0.0f;
    for (int s = 0; s < DSAMP; ++s) {
        float l = -1e30f;
        if (d < DDIM) {
            float u = u01((unsigned)(((s * NSENT + n) * 64 + d) * 2654435761u + 0x9e3779b9u));
            float g = -__logf(-__logf(u + 1e-10f) + 1e-10f);
            l = base + g;
        }
        sv[d] = l;
        __syncthreads();
        if (d == 0) {
            float m = -1e30f;
            for (int i = 0; i < DDIM; ++i) m = fmaxf(m, sv[i]);
            red = m;
        }
        __syncthreads();
        float e = (d < DDIM) ? __expf(l - red) : 0.0f;
        sv[d] = e;
        __syncthreads();
        if (d == 0) {
            float s2 = 0.0f;
            for (int i = 0; i < DDIM; ++i) s2 += sv[i];
            red = s2;
        }
        __syncthreads();
        if (d < DDIM) accp += e / red;
        __syncthreads();
    }
    float out = (d < DDIM) ? accp * (1.0f / DSAMP) : 0.0f;
    if (d < DDIM) sample_d[n * DDIM + d] = out;
    dfuse16[(size_t)n * DKPAD + d] = (_Float16)out;
}

// ------ gaussian reparam sample + softmax; writes cols 64..127 of fused A ------
__global__ void topic_sample_kernel(const float* __restrict__ zmu, const float* __restrict__ zlv,
                                    float* __restrict__ sample_t,
                                    _Float16* __restrict__ dfuse16) {  // [NS, 128]
    const int n = blockIdx.x, d = threadIdx.x;  // 64 threads
    __shared__ float sv[64];
    __shared__ float red;
    float st = -1e30f;
    if (d < KDIM) {
        unsigned idx = (unsigned)(n * KDIM + d);
        float u1 = u01(idx * 747796405u + 0x85ebca6bu);
        float u2 = u01(idx * 2891336453u + 0xc2b2ae35u);
        float eps = sqrtf(-2.0f * __logf(u1)) * __cosf(6.2831853f * u2);
        st = zmu[n * KDIM + d] + __expf(0.5f * zlv[n * KDIM + d]) * eps;
        sample_t[n * KDIM + d] = st;
    }
    sv[d] = st;
    __syncthreads();
    if (d == 0) {
        float m = -1e30f;
        for (int i = 0; i < KDIM; ++i) m = fmaxf(m, sv[i]);
        red = m;
    }
    __syncthreads();
    float e = (d < KDIM) ? __expf(st - red) : 0.0f;
    sv[d] = e;
    __syncthreads();
    if (d == 0) {
        float s2 = 0.0f;
        for (int i = 0; i < KDIM; ++i) s2 += sv[i];
        red = s2;
    }
    __syncthreads();
    dfuse16[(size_t)n * DKPAD + NPAD64 + d] = (_Float16)((d < KDIM) ? e / red : 0.0f);
}

// ---------------- fused last-turn memory read + score ----------------
__global__ __launch_bounds__(256)
void memory_score_kernel(const int* __restrict__ turn_lens,
                         const float* __restrict__ sample_d,
                         const float* __restrict__ sample_t,
                         const float* __restrict__ sent_emb,
                         const float* __restrict__ eraser_W, const float* __restrict__ eraser_b,
                         const float* __restrict__ update_W, const float* __restrict__ update_b,
                         const float* __restrict__ memory,   // [D+K, M]
                         const float* __restrict__ fc_W,     // [M]
                         float* __restrict__ score_out) {
    const int b = blockIdx.x;
    const int tid = threadIdx.x;  // 256 == MDIM
    __shared__ int sn;
    __shared__ float ms[DDIM + KDIM];
    __shared__ float sred[256];
    if (tid == 0) {
        int a = 0;
        for (int i = 0; i <= b; ++i) a += turn_lens[i];
        sn = a - 1;
    }
    __syncthreads();
    const int n = sn;
    if (tid < DDIM) ms[tid] = sample_d[n * DDIM + tid];
    else if (tid < DDIM + KDIM) ms[tid] = sample_t[n * KDIM + (tid - DDIM)];
    __syncthreads();

    const int m = tid;
    float ea = eraser_b[m], ua = update_b[m];
    for (int h = 0; h < HDIM; ++h) {
        float se = sent_emb[(size_t)n * HDIM + h];
        ea += se * eraser_W[(size_t)m * HDIM + h];
        ua += se * update_W[(size_t)m * HDIM + h];
    }
    const float er = 1.0f / (1.0f + __expf(-ea));
    const float up = tanhf(ua);
    float o = 0.0f;
    for (int d = 0; d < DDIM + KDIM; ++d) {
        float msd = ms[d];
        float mem = memory[d * MDIM + m] * (1.0f - msd * er) + msd * up;
        o += msd * mem;
    }
    sred[tid] = o * fc_W[m];
    __syncthreads();
    for (int s = 128; s > 0; s >>= 1) {
        if (tid < s) sred[tid] += sred[tid + s];
        __syncthreads();
    }
    if (tid == 0) score_out[b] = sred[0];
}

// ---------------- host-side orchestration ----------------
extern "C" void kernel_launch(void* const* d_in, const int* in_sizes, int n_in,
                              void* d_out, int out_size, void* d_ws, size_t ws_size,
                              hipStream_t stream) {
    const int*   word_id   = (const int*)  d_in[0];
    const float* sent_bow  = (const float*)d_in[1];
    const int*   sent_lens = (const int*)  d_in[2];
    const int*   turn_lens = (const int*)  d_in[3];
    const float* word_emb  = (const float*)d_in[4];
    const float* w_ih      = (const float*)d_in[5];
    const float* w_hh      = (const float*)d_in[6];
    const float* b_ih      = (const float*)d_in[7];
    const float* b_hh      = (const float*)d_in[8];
    const float* attn_W    = (const float*)d_in[9];
    const float* attn_b    = (const float*)d_in[10];
    const float* attn_v    = (const float*)d_in[11];
    const float* xenc_W    = (const float*)d_in[12];
    const float* xenc_b    = (const float*)d_in[13];
    const float* ctxenc_W  = (const float*)d_in[14];
    const float* ctxenc_b  = (const float*)d_in[15];
    const float* zmu_W     = (const float*)d_in[16];
    const float* zmu_b     = (const float*)d_in[17];
    const float* zlv_W     = (const float*)d_in[18];
    const float* zlv_b     = (const float*)d_in[19];
    const float* xdec_W    = (const float*)d_in[20];
    const float* xdec_b    = (const float*)d_in[21];
    const float* ctxdec_W  = (const float*)d_in[22];
    const float* ctxdec_b  = (const float*)d_in[23];
    const float* memory    = (const float*)d_in[24];
    const float* eraser_W  = (const float*)d_in[25];
    const float* eraser_b  = (const float*)d_in[26];
    const float* update_W  = (const float*)d_in[27];
    const float* update_b  = (const float*)d_in[28];
    const float* fc_W      = (const float*)d_in[29];
    (void)in_sizes; (void)n_in; (void)out_size; (void)ws_size;

    float* recon = (float*)d_out;                       // [NS, V]
    float* score = recon + (size_t)NSENT * VDIM;        // [B]

    // workspace layout (all 256B aligned)
    char* w = (char*)d_ws;
    size_t off = 0;
    auto take = [&](size_t bytes) -> char* {
        char* p = w + off;
        off += (bytes + 255) & ~(size_t)255;
        return p;
    };
    _Float16* bow16    = (_Float16*)take((size_t)NSENT * VDIM * 2);   // [1024,20000]
    _Float16* wemb16   = (_Float16*)take((size_t)VDIM * EPAD * 2);    // [20000,320]
    _Float16* wih16    = (_Float16*)take((size_t)3 * HDIM * EPAD * 2);
    _Float16* whh16    = (_Float16*)take((size_t)3 * HDIM * HDIM * 2);
    _Float16* attnW16  = (_Float16*)take((size_t)HDIM * HDIM * 2);
    _Float16* xenc16   = (_Float16*)take((size_t)NPAD64 * VDIM * 2);  // [64,20000]
    _Float16* ctxenc16 = (_Float16*)take((size_t)HDIM * VDIM * 2);    // [512,20000]
    _Float16* zmuW16   = (_Float16*)take((size_t)NPAD64 * HDIM * 2);
    _Float16* zlvW16   = (_Float16*)take((size_t)NPAD64 * HDIM * 2);
    _Float16* decW16   = (_Float16*)take((size_t)VPAD * DKPAD * 2);   // [20032,128] fused
    float*    bias2    = (float*)   take((size_t)VDIM * 4);           // xdec_b + ctxdec_b
    float*    h32      = (float*)   take((size_t)NSENT * HDIM * 4);
    _Float16* h16      = (_Float16*)take((size_t)NSENT * HDIM * 2);
    _Float16* outh16   = (_Float16*)take((size_t)NSENT * NWORD * HDIM * 2);
    float*    scores   = (float*)   take((size_t)NSENT * NWORD * 4);
    float*    semb     = (float*)   take((size_t)NSENT * HDIM * 4);
    float*    qd       = (float*)   take((size_t)NSENT * DDIM * 4);
    float*    ctx      = (float*)   take((size_t)NSENT * HDIM * 4);
    _Float16* ctx16    = (_Float16*)take((size_t)NSENT * HDIM * 2);
    float*    zmu      = (float*)   take((size_t)NSENT * KDIM * 4);
    float*    zlv      = (float*)   take((size_t)NSENT * KDIM * 4);
    float*    samp_d   = (float*)   take((size_t)NSENT * DDIM * 4);
    float*    samp_t   = (float*)   take((size_t)NSENT * KDIM * 4);
    _Float16* dfuse16  = (_Float16*)take((size_t)NSENT * DKPAD * 2);  // [1024,128] fused A

    auto cvt = [&](const float* s, _Float16* d, int sr, int sc, int dr, int dc,
                   int dld, int dcol0) {
        int n = dr * dc;
        cvt2d_f16_kernel<<<(n + 255) / 256, 256, 0, stream>>>(s, d, sr, sc, dr, dc, dld, dcol0);
    };
    // 1) f16 conversion / padding of all GEMM operands
    cvt(sent_bow, bow16,    NSENT, VDIM,  NSENT, VDIM,  VDIM, 0);
    cvt(word_emb, wemb16,   VDIM,  EDIM,  VDIM,  EPAD,  EPAD, 0);
    cvt(w_ih,     wih16,    3*HDIM, EDIM, 3*HDIM, EPAD, EPAD, 0);
    cvt(w_hh,     whh16,    3*HDIM, HDIM, 3*HDIM, HDIM, HDIM, 0);
    cvt(attn_W,   attnW16,  HDIM,  HDIM,  HDIM,  HDIM,  HDIM, 0);
    cvt(xenc_W,   xenc16,   DDIM,  VDIM,  NPAD64, VDIM, VDIM, 0);
    cvt(ctxenc_W, ctxenc16, HDIM,  VDIM,  HDIM,  VDIM,  VDIM, 0);
    cvt(zmu_W,    zmuW16,   KDIM,  HDIM,  NPAD64, HDIM, HDIM, 0);
    cvt(zlv_W,    zlvW16,   KDIM,  HDIM,  NPAD64, HDIM, HDIM, 0);
    cvt(xdec_W,   decW16,   VDIM,  DDIM,  VPAD,  NPAD64, DKPAD, 0);       // cols 0..63
    cvt(ctxdec_W, decW16,   VDIM,  KDIM,  VPAD,  NPAD64, DKPAD, NPAD64);  // cols 64..127
    add_bias_kernel<<<(VDIM + 255) / 256, 256, 0, stream>>>(xdec_b, ctxdec_b, bias2, VDIM);

    // 2) zero hidden state
    {
        int nh = NSENT * HDIM;
        zero_f32_kernel<<<(nh + 255) / 256, 256, 0, stream>>>(h32, nh);
        zero_f32_kernel<<<(nh / 2 + 255) / 256, 256, 0, stream>>>((float*)h16, nh / 2);
    }
    // 3) masked GRU, one fused WMMA step per timestep
    for (int t = 0; t < NWORD; ++t) {
        gru_step_kernel<<<dim3(HDIM / 64, NSENT / 64), 256, 0, stream>>>(
            word_id, wemb16, wih16, whh16, b_ih, b_hh, sent_lens,
            h32, h16, outh16, t);
    }
    // 4) attention scores + softmax pooling
    attn_score_kernel<<<NSENT * NWORD / 64, 256, 0, stream>>>(
        outh16, attnW16, attn_b, attn_v, scores);
    softmax_pool_kernel<<<NSENT, 128, 0, stream>>>(scores, sent_lens, outh16, semb);

    // 5) bag-of-words encoders (A = bow16, K = 20000)
    gemm_wmma_kernel<<<dim3(1, NSENT / 64), 256, 0, stream>>>(
        bow16, VDIM, xenc16, VDIM, xenc_b, qd, nullptr, DDIM,
        DDIM, VDIM, /*act*/0, /*nt*/0);
    gemm_wmma_kernel<<<dim3(HDIM / 64, NSENT / 64), 256, 0, stream>>>(
        bow16, VDIM, ctxenc16, VDIM, ctxenc_b, ctx, ctx16, HDIM,
        HDIM, VDIM, /*act tanh*/1, 0);

    // 6) gaussian head (A = ctx16)
    gemm_wmma_kernel<<<dim3(1, NSENT / 64), 256, 0, stream>>>(
        ctx16, HDIM, zmuW16, HDIM, zmu_b, zmu, nullptr, KDIM,
        KDIM, HDIM, 0, 0);
    gemm_wmma_kernel<<<dim3(1, NSENT / 64), 256, 0, stream>>>(
        ctx16, HDIM, zlvW16, HDIM, zlv_b, zlv, nullptr, KDIM,
        KDIM, HDIM, 0, 0);

    // 7) sampling (fills fused decoder A: [sample_d | softmax(sample_t)])
    gumbel_kernel<<<NSENT, 64, 0, stream>>>(qd, samp_d, dfuse16);
    topic_sample_kernel<<<NSENT, 64, 0, stream>>>(zmu, zlv, samp_t, dfuse16);

    // 8) fused decoder: recon = dfuse @ [xdec|ctxdec]^T + (xdec_b+ctxdec_b)
    //    single streaming pass, nontemporal stores
    gemm_wmma_kernel<<<dim3(VPAD / 64, NSENT / 64), 256, 0, stream>>>(
        dfuse16, DKPAD, decW16, DKPAD, bias2, recon, nullptr, VDIM,
        VDIM, DKPAD, 0, /*nt*/1);

    // 9) last-turn memory read + score
    memory_score_kernel<<<BDLG, 256, 0, stream>>>(
        turn_lens, samp_d, samp_t, semb, eraser_W, eraser_b,
        update_W, update_b, memory, fc_W, score);
}
